// MultiHeadAttention_18494129177321
// MI455X (gfx1250) — matrix operand order repaired and checked by reading, not compiled, not run on hardware
//
#include <hip/hip_runtime.h>
#include <hip/hip_bf16.h>
#include <cstdint>
#include <cstddef>

typedef __attribute__((ext_vector_type(16))) __bf16 v16bf;
typedef __attribute__((ext_vector_type(8)))  __bf16 v8bf;
typedef __attribute__((ext_vector_type(8)))  float  v8f;

#define B_     2
#define S_     2048
#define D_     1024
#define H_     16
#define DEPTH_ 64
#define LDS_PITCH (S_ + 4)   // pad to break 64-bank alignment on row-major LDS tile

// ---------------- WMMA fragment helpers (CDNA5 16x16x32 bf16) ----------------
// A (16x32, MxK): lane m (0..15) holds row m, K = 0..15 contiguous; lanes 16..31 hold
// the same rows with K = 16..31. Packed low-to-high in the 16-elt vector.
static __device__ __forceinline__ v16bf load_a_frag(const __bf16* __restrict__ base,
                                                    int ld, int lane) {
  const int row  = lane & 15;
  const int koff = (lane < 16) ? 0 : 16;
  const __bf16* p = base + (size_t)row * ld + koff;
  v8bf lo = *(const v8bf*)(p);
  v8bf hi = *(const v8bf*)(p + 8);
  v16bf r;
#pragma unroll
  for (int i = 0; i < 8; ++i) { r[i] = lo[i]; r[i + 8] = hi[i]; }
  return r;
}

// B (32x16, KxN) sourced from a transposed (N x K row-major) array:
// lane n (0..15) holds column n, K = 0..15 contiguous; lanes 16..31 hold K = 16..31.
static __device__ __forceinline__ v16bf load_b_frag(const __bf16* __restrict__ baseT,
                                                    int ld, int lane) {
  const int col  = lane & 15;
  const int koff = (lane < 16) ? 0 : 16;
  return *(const v16bf*)(baseT + (size_t)col * ld + koff);
}

static __device__ __forceinline__ v8f wmma_bf16(v16bf a, v16bf b, v8f c) {
  return __builtin_amdgcn_wmma_f32_16x16x32_bf16(false, a, false, b, (short)0, c,
                                                 false, false);
}

// CDNA5 async copy: one instruction per wave, 16B per lane, tracked by ASYNCcnt.
// VDST VGPR = LDS byte address (low 32 bits of the generic shared pointer),
// VADDR pair = 64-bit global address.
static __device__ __forceinline__ void async_copy_b128(uint32_t lds_byte,
                                                       const void* gaddr) {
  asm volatile("global_load_async_to_lds_b128 %0, %1, off"
               :: "v"(lds_byte), "v"((uint64_t)(uintptr_t)gaddr) : "memory");
}

// ---------------- conversion kernels ----------------
__global__ void f32_to_bf16_kernel(const float* __restrict__ in,
                                   __bf16* __restrict__ out, int n) {
  int i = blockIdx.x * blockDim.x + threadIdx.x;
  if (i < n) out[i] = (__bf16)in[i];
}

// Wt[n][k] = (bf16) W[k][n]   (K x N  ->  N x K)
__global__ void transpose_w_bf16_kernel(const float* __restrict__ W,
                                        __bf16* __restrict__ Wt, int K, int N) {
  int n = blockIdx.x * 32 + (threadIdx.x & 31);
  int k = blockIdx.y * 8 + (threadIdx.x >> 5);
  if (n < N && k < K) Wt[(size_t)n * K + k] = (__bf16)W[(size_t)k * N + n];
}

// ---------------- WMMA GEMM with async-LDS weight staging ----------------
// C = A(MxK) @ B(KxN) + bias.  A bf16 row-major, Bt bf16 NxK row-major.
// Block = 256 threads (8 waves) computing a 128M x 64N tile:
//   wave w owns rows [bm + 16w, +16), 4 accumulators over the 64 N columns.
// Per k-step the 64N x 32K weight slice (4 KB) is staged in LDS via
// global_load_async_to_lds_b128, double-buffered and overlapped with WMMA.
// mode 0: f32 row-major; mode 1: bf16 (B,H,S,64); mode 2: bf16 (B,H,64,S).
__global__ __launch_bounds__(256) void gemm_bf16_wmma(
    const __bf16* __restrict__ A, const __bf16* __restrict__ Bt,
    const float* __restrict__ bias, float* __restrict__ Cf,
    __bf16* __restrict__ Cb, int M, int N, int K, int mode) {
  __shared__ __align__(16) __bf16 lds_bt[2][64 * 32];

  const int t    = threadIdx.x;
  const int w    = t >> 5;
  const int lane = t & 31;
  const int bm   = blockIdx.x * 128;
  const int bn   = blockIdx.y * 64;

  // async-copy mapping: thread t moves one 16B chunk: row = t/4, chunk = t%4
  const int crow = t >> 2;
  const int celt = (t & 3) * 8;
  const uint32_t ldst0 = (uint32_t)(uintptr_t)&lds_bt[0][crow * 32 + celt];
  const uint32_t ldst1 = (uint32_t)(uintptr_t)&lds_bt[1][crow * 32 + celt];
  const __bf16* gsrc = Bt + (size_t)(bn + crow) * K + celt;

  const __bf16* pa = A + (size_t)(bm + w * 16) * K;

  v8f acc0 = {}, acc1 = {}, acc2 = {}, acc3 = {};
  const int nk = K >> 5;

  async_copy_b128(ldst0, gsrc);            // prime buffer 0 (k-step 0)
  for (int kk = 0; kk < nk; ++kk) {
    if (kk + 1 < nk) {                     // uniform branch
      async_copy_b128(((kk + 1) & 1) ? ldst1 : ldst0, gsrc + (kk + 1) * 32);
      asm volatile("s_wait_asynccnt 0x1" ::: "memory");  // buffer kk landed
    } else {
      asm volatile("s_wait_asynccnt 0x0" ::: "memory");
    }
    __syncthreads();

    __builtin_prefetch(pa + (kk + 1) * 32, 0, 0);
    v16bf a = load_a_frag(pa + kk * 32, K, lane);

    const __bf16* lb = &lds_bt[kk & 1][0];
    const int col  = lane & 15;
    const int koff = (lane < 16) ? 0 : 16;
    v16bf b0 = *(const v16bf*)(lb + (0 * 16 + col) * 32 + koff);
    v16bf b1 = *(const v16bf*)(lb + (1 * 16 + col) * 32 + koff);
    v16bf b2 = *(const v16bf*)(lb + (2 * 16 + col) * 32 + koff);
    v16bf b3 = *(const v16bf*)(lb + (3 * 16 + col) * 32 + koff);
    acc0 = wmma_bf16(a, b0, acc0);
    acc1 = wmma_bf16(a, b1, acc1);
    acc2 = wmma_bf16(a, b2, acc2);
    acc3 = wmma_bf16(a, b3, acc3);

    __syncthreads();                       // reads done before buffer reuse
  }

  const int col = lane & 15;
  const int mbv = (lane >> 4) << 3;
  v8f accs[4] = {acc0, acc1, acc2, acc3};
#pragma unroll
  for (int j = 0; j < 4; ++j) {
    const int n  = bn + j * 16 + col;
    const float bvp = bias[n];
#pragma unroll
    for (int v = 0; v < 8; ++v) {
      const int m = bm + w * 16 + mbv + v;
      const float val = accs[j][v] + bvp;
      if (mode == 0) {
        Cf[(size_t)m * N + n] = val;
      } else {
        const int bb = m >> 11, s = m & (S_ - 1);
        const int h = n >> 6, d = n & (DEPTH_ - 1);
        size_t idx;
        if (mode == 1) idx = (((size_t)(bb * H_ + h) * S_) + s) * DEPTH_ + d;
        else           idx = (((size_t)(bb * H_ + h) * DEPTH_) + d) * S_ + s;
        Cb[idx] = (__bf16)val;
      }
    }
  }
}

// ---------------- fused causal attention (per 16-row q tile) ----------------
// One block = (b, h, qt). 4 waves. Full 16 x S logits block in LDS (128 KB of
// the 320 KB WGP LDS), softmax in LDS, attn streamed to global, then P@V via WMMA.
__global__ __launch_bounds__(128) void attn_kernel(
    const __bf16* __restrict__ Qh, const __bf16* __restrict__ Kh,
    const __bf16* __restrict__ Vt, float* __restrict__ attn_out,
    __bf16* __restrict__ concat) {
  __shared__ float lds_p[16][LDS_PITCH];
  __shared__ float lds_red[16][8];
  __shared__ float lds_m[16];
  __shared__ float lds_inv[16];

  const int bid = blockIdx.x;
  const int qt  = bid & 127;         // S/16 = 128 q tiles
  const int h   = (bid >> 7) & (H_ - 1);
  const int b   = bid >> 11;
  const int t    = threadIdx.x;
  const int w    = t >> 5;           // wave 0..3
  const int lane = t & 31;

  const size_t bh = (size_t)(b * H_ + h);
  const __bf16* qbase  = Qh + (bh * S_ + (size_t)qt * 16) * DEPTH_;
  const __bf16* kbase0 = Kh + bh * S_ * DEPTH_;

  // Q A-fragments for depth chunks k=0..31 and k=32..63 (shared by all key tiles)
  v16bf aq0 = load_a_frag(qbase, DEPTH_, lane);
  v16bf aq1 = load_a_frag(qbase + 32, DEPTH_, lane);

  const int col = lane & 15;
  const int mb  = (lane >> 4) << 3;

  // ---- logits: S = Q K^T * 1/sqrt(64), causal mask, into LDS ----
  for (int kt = w; kt < (S_ >> 4); kt += 4) {
    if (kt > qt) {  // fully masked tile (wave-uniform branch, EXEC stays full)
#pragma unroll
      for (int v = 0; v < 8; ++v) lds_p[mb + v][kt * 16 + col] = -1.0e9f;
    } else {
      const __bf16* kb = kbase0 + (size_t)kt * 16 * DEPTH_;
      v16bf b0 = load_b_frag(kb, DEPTH_, lane);
      v16bf b1 = load_b_frag(kb + 32, DEPTH_, lane);
      v8f acc = {};
      acc = wmma_bf16(aq0, b0, acc);
      acc = wmma_bf16(aq1, b1, acc);
#pragma unroll
      for (int v = 0; v < 8; ++v) {
        const int r = mb + v;
        float val = acc[v] * 0.125f;                       // 1/sqrt(depth)
        if (kt == qt && col > r) val = -1.0e9f;            // diagonal tile mask
        lds_p[r][kt * 16 + col] = val;
      }
    }
  }
  __syncthreads();

  // ---- softmax over each of the 16 rows (8 threads per row) ----
  const int r = t >> 3, sub = t & 7;
  float mx = -3.0e38f;
  for (int c = sub; c < S_; c += 8) mx = fmaxf(mx, lds_p[r][c]);
  lds_red[r][sub] = mx;
  __syncthreads();
  if (sub == 0) {
    float mm = lds_red[r][0];
#pragma unroll
    for (int i = 1; i < 8; ++i) mm = fmaxf(mm, lds_red[r][i]);
    lds_m[r] = mm;
  }
  __syncthreads();
  const float mm = lds_m[r];
  float sum = 0.f;
  for (int c = sub; c < S_; c += 8) {
    float e = __expf(lds_p[r][c] - mm);
    lds_p[r][c] = e;
    sum += e;
  }
  lds_red[r][sub] = sum;
  __syncthreads();
  if (sub == 0) {
    float s = 0.f;
#pragma unroll
    for (int i = 0; i < 8; ++i) s += lds_red[r][i];
    lds_inv[r] = 1.0f / s;
  }
  __syncthreads();
  const float inv = lds_inv[r];
  float* arow = attn_out + (bh * S_ + (size_t)qt * 16 + r) * S_;
  for (int c = sub; c < S_; c += 8) {
    float p = lds_p[r][c] * inv;
    lds_p[r][c] = p;     // normalized P stays in LDS for the PV matmul
    arow[c] = p;         // required output: full attention matrix
  }
  __syncthreads();

  // ---- ctx = P (16 x S) @ V (S x 64): wave w owns d-columns [w*16, w*16+16) ----
  const __bf16* vb = Vt + (bh * DEPTH_ + (size_t)w * 16) * S_;
  v8f acc = {};
  const int kcmax = (qt * 16 + 15) >> 5;  // beyond this P is exactly 0
  for (int kc = 0; kc <= kcmax; ++kc) {
    const int ar   = lane & 15;
    const int koff = (lane < 16) ? 0 : 16;
    const int cb   = kc * 32;
    v16bf a;
#pragma unroll
    for (int i = 0; i < 8; ++i) {
      a[i]     = (__bf16)lds_p[ar][cb + koff + i];
      a[i + 8] = (__bf16)lds_p[ar][cb + koff + 8 + i];
    }
    v16bf bfr = load_b_frag(vb + cb, S_, lane);
    acc = wmma_bf16(a, bfr, acc);
  }
#pragma unroll
  for (int v = 0; v < 8; ++v) {
    const int rr = mb + v;
    const int s  = qt * 16 + rr;
    const int d  = w * 16 + col;
    concat[((size_t)b * S_ + s) * D_ + h * DEPTH_ + d] = (__bf16)acc[v];
  }
}

// ---------------- host orchestration ----------------
extern "C" void kernel_launch(void* const* d_in, const int* in_sizes, int n_in,
                              void* d_out, int out_size, void* d_ws, size_t ws_size,
                              hipStream_t stream) {
  (void)in_sizes; (void)n_in; (void)out_size; (void)ws_size;
  const float* v_in = (const float*)d_in[0];
  const float* k_in = (const float*)d_in[1];
  const float* q_in = (const float*)d_in[2];
  // d_in[3] is the causal mask; applied analytically in attn_kernel.
  const float* Wq = (const float*)d_in[4];
  const float* bq = (const float*)d_in[5];
  const float* Wk = (const float*)d_in[6];
  const float* bk = (const float*)d_in[7];
  const float* Wv = (const float*)d_in[8];
  const float* bv = (const float*)d_in[9];
  const float* Wo = (const float*)d_in[10];
  const float* bo = (const float*)d_in[11];

  const size_t XN  = (size_t)B_ * S_ * D_;           // 4,194,304 elements
  const size_t WN  = (size_t)D_ * D_;                // 1,048,576 elements
  const size_t HDN = (size_t)B_ * H_ * S_ * DEPTH_;  // == XN

  char* ws = (char*)d_ws;
  size_t off = 0;
  __bf16* xq  = (__bf16*)(ws + off); off += XN * 2;
  __bf16* xk  = (__bf16*)(ws + off); off += XN * 2;
  __bf16* xv  = (__bf16*)(ws + off); off += XN * 2;
  __bf16* Wqt = (__bf16*)(ws + off); off += WN * 2;
  __bf16* Wkt = (__bf16*)(ws + off); off += WN * 2;
  __bf16* Wvt = (__bf16*)(ws + off); off += WN * 2;
  __bf16* Wot = (__bf16*)(ws + off); off += WN * 2;
  __bf16* Qh  = (__bf16*)(ws + off); off += HDN * 2;
  __bf16* Kh  = (__bf16*)(ws + off); off += HDN * 2;
  __bf16* Vt  = (__bf16*)(ws + off); off += HDN * 2;
  __bf16* ccat= (__bf16*)(ws + off); off += XN * 2;   // ~64 MB total

  float* out_f    = (float*)d_out;
  float* attn_out = out_f + XN;   // outputs concatenated: [out | attn]

  // 1) input conversions
  f32_to_bf16_kernel<<<(int)(XN / 256), 256, 0, stream>>>(q_in, xq, (int)XN);
  f32_to_bf16_kernel<<<(int)(XN / 256), 256, 0, stream>>>(k_in, xk, (int)XN);
  f32_to_bf16_kernel<<<(int)(XN / 256), 256, 0, stream>>>(v_in, xv, (int)XN);
  dim3 tg(D_ / 32, D_ / 8);
  transpose_w_bf16_kernel<<<tg, 256, 0, stream>>>(Wq, Wqt, D_, D_);
  transpose_w_bf16_kernel<<<tg, 256, 0, stream>>>(Wk, Wkt, D_, D_);
  transpose_w_bf16_kernel<<<tg, 256, 0, stream>>>(Wv, Wvt, D_, D_);
  transpose_w_bf16_kernel<<<tg, 256, 0, stream>>>(Wo, Wot, D_, D_);

  // 2) projections: (4096x1024)@(1024x1024); block tile 128M x 64N
  const int M = B_ * S_;
  dim3 gg(M / 128, D_ / 64);   // (32, 16)
  gemm_bf16_wmma<<<gg, 256, 0, stream>>>(xq, Wqt, bq, nullptr, Qh, M, D_, D_, 1);
  gemm_bf16_wmma<<<gg, 256, 0, stream>>>(xk, Wkt, bk, nullptr, Kh, M, D_, D_, 1);
  gemm_bf16_wmma<<<gg, 256, 0, stream>>>(xv, Wvt, bv, nullptr, Vt, M, D_, D_, 2);

  // 3) fused causal attention: one block per (b,h,16-row q tile)
  attn_kernel<<<B_ * H_ * (S_ / 16), 128, 0, stream>>>(Qh, Kh, Vt, attn_out, ccat);

  // 4) output projection -> f32 d_out
  gemm_bf16_wmma<<<gg, 256, 0, stream>>>(ccat, Wot, bo, out_f, nullptr, M, D_, D_, 0);
}
